// CustomRNN_39968965656997
// MI455X (gfx1250) — matrix operand order repaired and checked by reading, not compiled, not run on hardware
//
#include <hip/hip_runtime.h>
#include <hip/hip_bf16.h>

// ---------------------------------------------------------------------------
// 2-layer tanh RNN for MI455X (gfx1250, wave32, WMMA):
//   Phase A (parallel):  Ain = bf16(x) @ W_in + b_in                   [32768,512]
//                        P0  = Ain @ W_i2h0 + b_i2h0   (fp32)          [32768,512]
//   Phase B (serial)  :  recurrence split over 8 workgroups (64 N-cols each);
//                        h-state exchanged via L2 + global split-barrier,
//                        staged into LDS for ds_load A-fragments.
//   Phase C (parallel):  out = H1 @ W_out + b_out      (fp32)          [32768,256]
// All GEMMs: v_wmma_f32_16x16x32_bf16, fp32 accumulation. Weights pre-
// transposed to [N,K] bf16 so A/B fragments are contiguous 16B loads.
// Weight pointers are laundered through empty inline-asm inside the time loop
// so LICM cannot hoist (and then spill) the L2-resident weight fragments.
// ---------------------------------------------------------------------------

typedef __attribute__((ext_vector_type(16))) __bf16 v16bf;
typedef __attribute__((ext_vector_type(8)))  __bf16 v8bf;
typedef __attribute__((ext_vector_type(8)))  float  v8f;

#define RNN_I 256
#define RNN_H 512
#define RNN_O 256
#define RNN_B 64
#define RNN_S 512
#define RNN_C 8          // workgroups cooperating on the serial recurrence

static __device__ __forceinline__ v16bf frag16(const __bf16* p0, const __bf16* p1) {
  v8bf lo = *(const v8bf*)p0;
  v8bf hi = *(const v8bf*)p1;
  return __builtin_shufflevector(lo, hi, 0,1,2,3,4,5,6,7,8,9,10,11,12,13,14,15);
}

static __device__ __forceinline__ v8f wmma_bf16(v16bf a, v16bf b, v8f c) {
  return __builtin_amdgcn_wmma_f32_16x16x32_bf16(false, a, false, b, (short)0, c,
                                                 false, false);
}

// ---- elementwise helpers ---------------------------------------------------

__global__ void k_cast_bf16(const float* __restrict__ in, __bf16* __restrict__ out,
                            long n) {
  long idx = (long)blockIdx.x * 256 + threadIdx.x;
  if (idx < n) out[idx] = (__bf16)in[idx];
}

// W: [K,N] fp32 row-major  ->  WT: [N,K] bf16 row-major
__global__ void k_transpose_bf16(const float* __restrict__ W, __bf16* __restrict__ WT,
                                 int K, int N) {
  long idx = (long)blockIdx.x * 256 + threadIdx.x;
  if (idx < (long)K * N) {
    int k = (int)(idx / N);
    int n = (int)(idx % N);
    WT[(long)n * K + k] = (__bf16)W[idx];
  }
}

__global__ void k_bar_init(unsigned* bar) { bar[0] = 0u; bar[1] = 0u; }

// ---- batched GEMM: C[M,N] = A[M,K] * BT[N,K]^T + bias ----------------------

template <bool OUT_BF16>
__global__ __launch_bounds__(256) void k_gemm_tn(const __bf16* __restrict__ A,
                                                 const __bf16* __restrict__ BT,
                                                 const float* __restrict__ bias,
                                                 void* __restrict__ Cout,
                                                 int M, int N, int K) {
  const int lane = threadIdx.x & 31;
  const int wave = threadIdx.x >> 5;                 // 0..7
  const int lr   = lane & 15;
  const int lh   = lane >> 4;
  const int m0   = blockIdx.x * 128 + wave * 16;
  const int nb   = blockIdx.y * 64;

  v8f acc[4];
#pragma unroll
  for (int j = 0; j < 4; ++j) acc[j] = (v8f){0.f,0.f,0.f,0.f,0.f,0.f,0.f,0.f};

  const __bf16* arow = A + (size_t)(m0 + lr) * K;
  for (int k0 = 0; k0 < K; k0 += 32) {
    v16bf af = frag16(arow + k0 + lh * 8, arow + k0 + 16 + lh * 8);
#pragma unroll
    for (int j = 0; j < 4; ++j) {
      const __bf16* bp = BT + (size_t)(nb + j * 16 + lr) * K + k0 + lh * 8;
      acc[j] = wmma_bf16(af, frag16(bp, bp + 16), acc[j]);
    }
  }

#pragma unroll
  for (int j = 0; j < 4; ++j) {
    int   n  = nb + j * 16 + lr;
    float bv = bias[n];
#pragma unroll
    for (int r = 0; r < 8; ++r) {
      int   m = m0 + lh * 8 + r;
      float v = acc[j][r] + bv;
      if (OUT_BF16) ((__bf16*)Cout)[(size_t)m * N + n] = (__bf16)v;
      else          ((float*)Cout)[(size_t)m * N + n]  = v;
    }
  }
}

// ---- device-wide split barrier (sense via generation counter) --------------

static __device__ __forceinline__ void gbarrier(unsigned* cnt, unsigned* gen,
                                                unsigned nblk) {
  __syncthreads();
  if (threadIdx.x == 0) {
    __threadfence();  // release my stores to device scope
    unsigned g = __hip_atomic_load(gen, __ATOMIC_ACQUIRE, __HIP_MEMORY_SCOPE_AGENT);
    unsigned a = __hip_atomic_fetch_add(cnt, 1u, __ATOMIC_ACQ_REL,
                                        __HIP_MEMORY_SCOPE_AGENT);
    if (a == nblk - 1u) {
      __hip_atomic_store(cnt, 0u, __ATOMIC_RELAXED, __HIP_MEMORY_SCOPE_AGENT);
      __hip_atomic_store(gen, g + 1u, __ATOMIC_RELEASE, __HIP_MEMORY_SCOPE_AGENT);
    } else {
      while (__hip_atomic_load(gen, __ATOMIC_ACQUIRE, __HIP_MEMORY_SCOPE_AGENT) == g)
        __builtin_amdgcn_s_sleep(2);
    }
    __threadfence();  // acquire peers' stores
  }
  __syncthreads();
}

// ---- serial recurrence: RNN_C cooperating workgroups -----------------------
// WG c owns N-columns [c*64, c*64+64) (4 n-tiles). 512 threads = 16 waves;
// wave w -> (tm = w&3, tn = w>>2): exactly one 16x16 accumulator tile.
// Full h0/h1 staged in LDS (bf16) for ds_load A-fragments; slices exchanged
// through global bf16 buffers + gbarrier each phase.

__global__ __launch_bounds__(512) void k_rnn_recurrent(
    const __bf16* __restrict__ Wh0T, const __bf16* __restrict__ Wi1T,
    const __bf16* __restrict__ Wh1T,
    const float* __restrict__ b_i2h,   // [L,H]
    const float* __restrict__ b_h2h,   // [L,H]
    const float* __restrict__ P0,      // [B*S,H], row b*S+t
    const float* __restrict__ h_prev,  // [L,B,H]
    __bf16* __restrict__ H1,           // [B*S,H], row b*S+t
    float* __restrict__ h_out,         // [L,B,H]
    __bf16* __restrict__ g_h0,         // [B,H] exchange buffer
    __bf16* __restrict__ g_h1,         // [B,H] exchange buffer
    unsigned* __restrict__ bar)        // {count, generation}
{
  extern __shared__ __bf16 lds[];
  __bf16* h0 = lds;                    // [B,H]
  __bf16* h1 = lds + RNN_B * RNN_H;    // [B,H]

  const int tid  = threadIdx.x;
  const int lane = tid & 31;
  const int wave = tid >> 5;           // 0..15
  const int lr   = lane & 15;
  const int lh   = lane >> 4;
  const int tm   = wave & 3;           // m-tile 0..3
  const int n    = blockIdx.x * 64 + (wave >> 2) * 16 + lr;
  const int mrow = tm * 16 + lh * 8;   // first of 8 output rows for this lane

  // init: every WG builds full bf16 state in its own LDS
  for (int idx = tid; idx < RNN_B * RNN_H; idx += 512) {
    h0[idx] = (__bf16)h_prev[idx];
    h1[idx] = (__bf16)h_prev[RNN_B * RNN_H + idx];
  }
  __syncthreads();

  const float bs0 = b_h2h[n];
  const float bs1 = b_i2h[RNN_H + n] + b_h2h[RNN_H + n];
  const __bf16* w0row  = Wh0T + (size_t)n * RNN_H;
  const __bf16* wi1row = Wi1T + (size_t)n * RNN_H;
  const __bf16* wh1row = Wh1T + (size_t)n * RNN_H;

  const int NV = RNN_B * RNN_H / 8;    // 16B-vector count for state copies

  for (int t = 0; t < RNN_S; ++t) {
    // Launder a zero offset so the compiler cannot prove the weight loads
    // loop-invariant (LICM would hoist ~48 fragments and spill to scratch).
    int woff = 0;
    asm volatile("" : "+v"(woff));
    const __bf16* w0p  = w0row  + woff;
    const __bf16* wi1p = wi1row + woff;
    const __bf16* wh1p = wh1row + woff;

    if (t + 1 < RNN_S) {
#pragma unroll
      for (int r = 0; r < 8; ++r)
        __builtin_prefetch(
            &P0[((size_t)(mrow + r) * RNN_S + (t + 1)) * RNN_H + n], 0, 3);
    }

    // ---- layer 0: acc = h0 @ W_h2h0 (K=512) ----
    v8f acc = (v8f){0.f,0.f,0.f,0.f,0.f,0.f,0.f,0.f};
    for (int k0 = 0; k0 < RNN_H; k0 += 32) {
      const __bf16* bp = w0p + k0 + lh * 8;
      const __bf16* ap = h0 + (size_t)(tm * 16 + lr) * RNN_H + k0 + lh * 8;
      acc = wmma_bf16(frag16(ap, ap + 16), frag16(bp, bp + 16), acc);
    }
#pragma unroll
    for (int r = 0; r < 8; ++r) {
      int   m = mrow + r;
      float v = tanhf(acc[r] + bs0 + P0[((size_t)m * RNN_S + t) * RNN_H + n]);
      g_h0[(size_t)m * RNN_H + n] = (__bf16)v;
    }
    gbarrier(bar, bar + 1, RNN_C);

    // pull full h0_new into LDS
    for (int idx = tid; idx < NV; idx += 512)
      ((v8bf*)h0)[idx] = ((const v8bf*)g_h0)[idx];
    __syncthreads();

    // ---- layer 1: acc = h0_new @ W_i2h1 + h1 @ W_h2h1 ----
    acc = (v8f){0.f,0.f,0.f,0.f,0.f,0.f,0.f,0.f};
    for (int k0 = 0; k0 < RNN_H; k0 += 32) {
      const __bf16* bpi = wi1p + k0 + lh * 8;
      const __bf16* bph = wh1p + k0 + lh * 8;
      const __bf16* a0  = h0 + (size_t)(tm * 16 + lr) * RNN_H + k0 + lh * 8;
      const __bf16* a1  = h1 + (size_t)(tm * 16 + lr) * RNN_H + k0 + lh * 8;
      acc = wmma_bf16(frag16(a0, a0 + 16), frag16(bpi, bpi + 16), acc);
      acc = wmma_bf16(frag16(a1, a1 + 16), frag16(bph, bph + 16), acc);
    }
#pragma unroll
    for (int r = 0; r < 8; ++r) {
      int    m  = mrow + r;
      float  v  = tanhf(acc[r] + bs1);
      __bf16 bv = (__bf16)v;
      g_h1[(size_t)m * RNN_H + n]              = bv;
      H1[((size_t)m * RNN_S + t) * RNN_H + n]  = bv;
    }
    gbarrier(bar, bar + 1, RNN_C);

    // pull full h1_new into LDS
    for (int idx = tid; idx < NV; idx += 512)
      ((v8bf*)h1)[idx] = ((const v8bf*)g_h1)[idx];
    __syncthreads();
  }

  // every WG holds the full final state; WG 0 emits h_t (fp32)
  if (blockIdx.x == 0) {
    for (int idx = tid; idx < RNN_B * RNN_H; idx += 512) {
      h_out[idx]                 = (float)h0[idx];
      h_out[RNN_B * RNN_H + idx] = (float)h1[idx];
    }
  }
}

// ---------------------------------------------------------------------------

extern "C" void kernel_launch(void* const* d_in, const int* in_sizes, int n_in,
                              void* d_out, int out_size, void* d_ws, size_t ws_size,
                              hipStream_t stream) {
  const float* x      = (const float*)d_in[0];   // [B,S,I]
  const float* h_prev = (const float*)d_in[1];   // [L,B,H]
  const float* W_in   = (const float*)d_in[2];   // [I,H]
  const float* b_in   = (const float*)d_in[3];   // [H]
  const float* W_i2h  = (const float*)d_in[4];   // [L,H,H]
  const float* b_i2h  = (const float*)d_in[5];   // [L,H]
  const float* W_h2h  = (const float*)d_in[6];   // [L,H,H]
  const float* b_h2h  = (const float*)d_in[7];   // [L,H]
  const float* W_out  = (const float*)d_in[8];   // [H,O]
  const float* b_out  = (const float*)d_in[9];   // [O]
  float* out = (float*)d_out;                    // [B,S,O] ++ [L,B,H]

  char*  ws  = (char*)d_ws;
  size_t off = 0;
  auto take = [&](size_t bytes) -> void* {
    void* p = ws + off;
    off += (bytes + 255) & ~(size_t)255;
    return p;
  };

  __bf16*   xb    = (__bf16*)take((size_t)RNN_B * RNN_S * RNN_I * 2);
  __bf16*   WinT  = (__bf16*)take((size_t)RNN_H * RNN_I * 2);
  __bf16*   Wi0T  = (__bf16*)take((size_t)RNN_H * RNN_H * 2);
  __bf16*   Wh0T  = (__bf16*)take((size_t)RNN_H * RNN_H * 2);
  __bf16*   Wi1T  = (__bf16*)take((size_t)RNN_H * RNN_H * 2);
  __bf16*   Wh1T  = (__bf16*)take((size_t)RNN_H * RNN_H * 2);
  __bf16*   WoutT = (__bf16*)take((size_t)RNN_O * RNN_H * 2);
  __bf16*   Ain   = (__bf16*)take((size_t)RNN_B * RNN_S * RNN_H * 2);
  float*    P0    = (float*) take((size_t)RNN_B * RNN_S * RNN_H * 4);
  __bf16*   H1    = (__bf16*)take((size_t)RNN_B * RNN_S * RNN_H * 2);
  __bf16*   g_h0  = (__bf16*)take((size_t)RNN_B * RNN_H * 2);
  __bf16*   g_h1  = (__bf16*)take((size_t)RNN_B * RNN_H * 2);
  unsigned* bar   = (unsigned*)take(256);

  // ---- precision / layout prep (parallel, tiny) ----
  long nx = (long)RNN_B * RNN_S * RNN_I;
  k_cast_bf16<<<dim3((unsigned)((nx + 255) / 256)), dim3(256), 0, stream>>>(x, xb, nx);

  long whh = (long)RNN_H * RNN_H;
  k_transpose_bf16<<<dim3((unsigned)(((long)RNN_I * RNN_H + 255) / 256)), dim3(256), 0, stream>>>(
      W_in, WinT, RNN_I, RNN_H);
  k_transpose_bf16<<<dim3((unsigned)((whh + 255) / 256)), dim3(256), 0, stream>>>(
      W_i2h, Wi0T, RNN_H, RNN_H);
  k_transpose_bf16<<<dim3((unsigned)((whh + 255) / 256)), dim3(256), 0, stream>>>(
      W_h2h, Wh0T, RNN_H, RNN_H);
  k_transpose_bf16<<<dim3((unsigned)((whh + 255) / 256)), dim3(256), 0, stream>>>(
      W_i2h + (size_t)RNN_H * RNN_H, Wi1T, RNN_H, RNN_H);
  k_transpose_bf16<<<dim3((unsigned)((whh + 255) / 256)), dim3(256), 0, stream>>>(
      W_h2h + (size_t)RNN_H * RNN_H, Wh1T, RNN_H, RNN_H);
  k_transpose_bf16<<<dim3((unsigned)(((long)RNN_H * RNN_O + 255) / 256)), dim3(256), 0, stream>>>(
      W_out, WoutT, RNN_H, RNN_O);
  k_bar_init<<<dim3(1), dim3(1), 0, stream>>>(bar);

  const int M = RNN_B * RNN_S;  // 32768 rows

  // ---- phase A: input-dependent GEMMs, fully parallel ----
  dim3 blk(256);
  dim3 ga(M / 128, RNN_H / 64);
  k_gemm_tn<true ><<<ga, blk, 0, stream>>>(xb,  WinT, b_in,  Ain, M, RNN_H, RNN_I);
  k_gemm_tn<false><<<ga, blk, 0, stream>>>(Ain, Wi0T, b_i2h, P0,  M, RNN_H, RNN_H);

  // ---- phase B: serial recurrence, 8 cooperating workgroups ----
  size_t lds_bytes = (size_t)2 * RNN_B * RNN_H * sizeof(__bf16);  // 128 KB
  k_rnn_recurrent<<<dim3(RNN_C), dim3(512), lds_bytes, stream>>>(
      Wh0T, Wi1T, Wh1T, b_i2h, b_h2h, P0, h_prev, H1,
      out + (size_t)RNN_B * RNN_S * RNN_O, g_h0, g_h1, bar);

  // ---- phase C: output projection, fully parallel ----
  dim3 gc(M / 128, RNN_O / 64);
  k_gemm_tn<false><<<gc, blk, 0, stream>>>(H1, WoutT, b_out, out, M, RNN_O, RNN_H);
}
